// MultiHeadAttention_63299228009434
// MI455X (gfx1250) — matrix operand order repaired and checked by reading, not compiled, not run on hardware
//
#include <hip/hip_runtime.h>
#include <hip/hip_bf16.h>
#include <math.h>

// Problem constants
#define HEADS 16
#define DKDIM 64
#define BATCH 2
#define SEQ   2048
#define DM    1024   // d_model = HEADS*DKDIM

typedef __bf16 bf16;
typedef __attribute__((ext_vector_type(16))) __bf16 v16bf;
typedef __attribute__((ext_vector_type(8)))  float  v8f;
typedef __attribute__((ext_vector_type(8)))  __bf16 bf16x8;
typedef __attribute__((ext_vector_type(4)))  __bf16 bf16x4;
typedef __attribute__((ext_vector_type(4)))  float  f32x4;
typedef __attribute__((ext_vector_type(4)))  unsigned int u32x4;
typedef __attribute__((ext_vector_type(8)))  int i32x8;
typedef __attribute__((ext_vector_type(4)))  int i32x4;

__device__ __forceinline__ v8f wmma_bf16(v16bf a, v16bf b, v8f c) {
  // D = A(16x32 bf16) x B(32x16 bf16) + C(16x16 f32)
  return __builtin_amdgcn_wmma_f32_16x16x32_bf16(
      /*neg_a=*/false, a, /*neg_b=*/false, b,
      /*c_mod=*/(short)0, c, /*reuse_a=*/false, /*reuse_b=*/false);
}

// A-operand fragment: LDS tile is [M rows][K cols] row-major (stride in bf16).
// A 16x32 layout (contiguous, matching B): lanes 0-15 row M=r hold K=0..15;
// lanes 16-31 row M=r hold K=16..31.
__device__ __forceinline__ v16bf load_frag_a(const bf16* base, int stride, int r, int h) {
  const bf16* row = base + r * stride + h * 16;
  v16bf f;
#pragma unroll
  for (int j = 0; j < 16; ++j) f[j] = row[j];
  return f;
}

// B-operand fragment: LDS tile is [N rows][K cols] row-major (i.e. B^T).
// B 32x16 in contiguous k order (matching A): lane = column N; lanes 0-15 hold K=0..15 in
// v0..v7, lanes 16-31 hold K=16..31.
__device__ __forceinline__ v16bf load_frag_b(const bf16* base, int stride, int r, int h) {
  const bf16* row = base + r * stride + h * 16;
  v16bf f;
#pragma unroll
  for (int j = 0; j < 16; ++j) f[j] = row[j];
  return f;
}

// ---------------------------------------------------------------------------
// TDM: 2D tile load Global -> LDS via Tensor Data Mover (ISA ch.8 D# layout).
// Row = 64 bf16 (128B = 32 DWORDs); LDS padded to stride 72 bf16 via
// pad_interval=4 (every 32 DWORDs) + pad_amount=3 (4 DWORDs = 16B).
// All operands block-uniform -> SGPR groups.  Tracked by TENSORcnt.
// (6-arg clang-23 builtin form: g0, g1, g2, g3, extra, cpol)
// ---------------------------------------------------------------------------
__device__ __forceinline__ void tdm_load_tile64(unsigned lds_addr, const void* gptr,
                                                unsigned tile_rows) {
  unsigned long long ga = (unsigned long long)(uintptr_t)gptr;
  u32x4 g0;
  g0[0] = 1u;                                        // count=1 (valid), user mode
  g0[1] = lds_addr;                                  // LDS byte address
  g0[2] = (unsigned)(ga & 0xffffffffu);              // global_addr[31:0]
  g0[3] = (unsigned)((ga >> 32) & 0x01ffffffu)       // global_addr[56:32]
          | (2u << 30);                              // type=2 ("image")
  i32x8 g1;
  g1[0] = (1 << 16)                                  // data_size=1 -> 2 bytes
        | (1 << 20)                                  // pad_enable
        | (4 << 22)                                  // pad_interval: 32 DWORDs
        | (3 << 25);                                 // pad_amount: 4 DWORDs
  g1[1] = (int)(64u << 16);                          // tensor_dim0 = 64 elems
  g1[2] = (int)(tile_rows << 16);                    // tensor_dim1 = rows
  g1[3] = (int)(64u << 16);                          // tile_dim0 = 64
  g1[4] = (int)(tile_rows & 0xffffu);                // tile_dim1 = rows
  g1[5] = 64;                                        // tensor_dim0_stride = 64
  g1[6] = 0;
  g1[7] = 0;
  i32x4 z4 = {};
  i32x8 z8 = {};
  __builtin_amdgcn_tensor_load_to_lds(g0, g1, z4, z4, z8, 0);
}

// ---------------------------------------------------------------------------
// Kernel 1: convert Wq and Wo (f32, [K][N] row-major) to bf16 transposed [N][K]
// ---------------------------------------------------------------------------
__global__ __launch_bounds__(256) void mha_convert_weights(
    const float* __restrict__ Wq, const float* __restrict__ Wo,
    bf16* __restrict__ Wqt, bf16* __restrict__ Wot) {
  int idx = blockIdx.x * 256 + threadIdx.x;     // 0 .. 2*1024*1024-1
  int which = idx >> 20;
  int rem = idx & ((1 << 20) - 1);
  int n = rem >> 10;
  int k = rem & 1023;
  const float* src = which ? Wo : Wq;
  bf16* dst = which ? Wot : Wqt;
  dst[rem] = (bf16)src[k * DM + n];             // dst[n][k] = src[k][n]
}

// ---------------------------------------------------------------------------
// Kernel 2: projection GEMM  P = X(f32) @ Wq(bf16,T) + bq, scatter to
// [H][B][S][64] bf16.  grid = (DM/128, M/128, 3), 256 threads (8 waves).
// ---------------------------------------------------------------------------
__global__ __launch_bounds__(256) void mha_proj_kernel(
    const float* __restrict__ Xq, const float* __restrict__ Xk,
    const float* __restrict__ Xv, const bf16* __restrict__ Wqt,
    const float* __restrict__ bq, bf16* __restrict__ Qp,
    bf16* __restrict__ Kp, bf16* __restrict__ Vp) {
  __shared__ bf16 As[128 * 72];   // [M=128][K=64], stride 72 (pad)
  __shared__ bf16 Bs[128 * 72];   // [N=128][K=64]

  const int tid = threadIdx.x;
  const int lane = tid & 31, w = tid >> 5;
  const int r = lane & 15, h = lane >> 4;
  const int wm = w >> 1, wn = w & 1;            // wave tile 32x64
  const int n0 = blockIdx.x * 128;
  const int m0 = blockIdx.y * 128;
  const int sel = blockIdx.z;
  const float* X = (sel == 0) ? Xq : (sel == 1) ? Xk : Xv;
  bf16* P = (sel == 0) ? Qp : (sel == 1) ? Kp : Vp;

  v8f acc[2][4] = {};

  for (int k0 = 0; k0 < DM; k0 += 64) {
    if (k0 + 64 < DM) {
      __builtin_prefetch(X + (m0 + (tid >> 1)) * DM + k0 + 64, 0, 1);
      __builtin_prefetch(Wqt + (n0 + (tid >> 1)) * DM + k0 + 64, 0, 1);
    }
    // A tile: 128x64 f32 -> bf16 LDS (2048 float4 chunks, 8 per thread)
#pragma unroll
    for (int i = 0; i < 8; ++i) {
      int c = tid + i * 256;
      int row = c >> 4, cc = (c & 15) * 4;
      f32x4 v = *(const f32x4*)(X + (size_t)(m0 + row) * DM + k0 + cc);
      bf16x4 bv;
#pragma unroll
      for (int e = 0; e < 4; ++e) bv[e] = (bf16)v[e];
      *(bf16x4*)(As + row * 72 + cc) = bv;
    }
    // B tile: 128x64 bf16 (1024 chunks of 8, 4 per thread)
#pragma unroll
    for (int i = 0; i < 4; ++i) {
      int c = tid + i * 256;
      int row = c >> 3, cc = (c & 7) * 8;
      *(bf16x8*)(Bs + row * 72 + cc) =
          *(const bf16x8*)(Wqt + (size_t)(n0 + row) * DM + k0 + cc);
    }
    __syncthreads();

#pragma unroll
    for (int kk = 0; kk < 64; kk += 32) {
      v16bf a0 = load_frag_a(As + (wm * 32 + 0)  * 72 + kk, 72, r, h);
      v16bf a1 = load_frag_a(As + (wm * 32 + 16) * 72 + kk, 72, r, h);
#pragma unroll
      for (int fn = 0; fn < 4; ++fn) {
        v16bf bb = load_frag_b(Bs + (wn * 64 + fn * 16) * 72 + kk, 72, r, h);
        acc[0][fn] = wmma_bf16(a0, bb, acc[0][fn]);
        acc[1][fn] = wmma_bf16(a1, bb, acc[1][fn]);
      }
    }
    __syncthreads();
  }

  // Epilogue: +bias, scatter to [H][B][S][64]
#pragma unroll
  for (int fm = 0; fm < 2; ++fm) {
#pragma unroll
    for (int fn = 0; fn < 4; ++fn) {
      int n = n0 + wn * 64 + fn * 16 + r;
      int hh = n >> 6, d = n & 63;
      float bias = bq[n];
#pragma unroll
      for (int g = 0; g < 8; ++g) {
        int m = m0 + wm * 32 + fm * 16 + g + 8 * h;
        int b = m >> 11, s = m & 2047;
        float val = acc[fm][fn][g] + bias;
        P[(((size_t)(hh * BATCH + b) * SEQ) + s) * DKDIM + d] = (bf16)val;
      }
    }
  }
}

// ---------------------------------------------------------------------------
// Kernel 3: flash attention.  grid = (SEQ/64, H*B), 128 threads (4 waves).
// Dynamic LDS layout (bytes):
//   Ss  f32 [64][129]  @ 0      (33024)
//   Os  f32 [64][65]   @ 33024  (16640)
//   m   f32 [64]       @ 49664
//   l   f32 [64]       @ 49920
//   Qs  bf16 [64][72]  @ 50176  (9216)   <- TDM-loaded (padded stride)
//   Ks  bf16 [128][72] @ 59392  (18432)  <- TDM-loaded (padded stride)
//   Vt  bf16 [64][132] @ 77824  (16896)  (V transposed: [dk][kv])
//   Ps  bf16 [64][132] @ 94720  (16896)
// total 111616 B (LDS is 320KB/WGP on CDNA5)
// ---------------------------------------------------------------------------
#define FLASH_LDS_BYTES 111616
#define QS_OFF 50176
#define KS_OFF 59392

__global__ __launch_bounds__(128) void mha_flash_kernel(
    const bf16* __restrict__ Qp, const bf16* __restrict__ Kp,
    const bf16* __restrict__ Vp, bf16* __restrict__ Ob) {
  extern __shared__ char smem[];
  float* Ss    = (float*)(smem);
  float* Os    = (float*)(smem + 33024);
  float* m_run = (float*)(smem + 49664);
  float* l_run = (float*)(smem + 49920);
  bf16*  Qs    = (bf16*)(smem + QS_OFF);
  bf16*  Ks    = (bf16*)(smem + KS_OFF);
  bf16*  Vt    = (bf16*)(smem + 77824);
  bf16*  Ps    = (bf16*)(smem + 94720);

  const int tid = threadIdx.x;
  const int lane = tid & 31, w = tid >> 5;
  const int r = lane & 15, h = lane >> 4;
  const int s0 = blockIdx.x * 64;
  const int pair = blockIdx.y;                 // h*B + b
  const bf16* Qb = Qp + (size_t)pair * SEQ * DKDIM;
  const bf16* Kb = Kp + (size_t)pair * SEQ * DKDIM;
  const bf16* Vb = Vp + (size_t)pair * SEQ * DKDIM;
  const float inv_scale = 0.022097086912079608f;   // 1/sqrt(2048) (ref scales by seq_len)

  // Flat LDS addresses: low 32 bits of the generic pointer are the LDS offset.
  const unsigned lds_base = (unsigned)(uintptr_t)smem;

  // Q tile 64x64 via TDM (wave 0 issues; TDM ignores EXEC, 1 issue per wave)
  if (w == 0) tdm_load_tile64(lds_base + QS_OFF, Qb + (size_t)s0 * DKDIM, 64);
  for (int i = tid; i < 64 * 65; i += 128) Os[i] = 0.0f;
  if (tid < 64) { m_run[tid] = -3.0e38f; l_run[tid] = 0.0f; }
  if (w == 0) __builtin_amdgcn_s_wait_tensorcnt(0);
  __syncthreads();

  const int wm = w >> 1, wn = w & 1;           // stage A wave tile 32x64

  for (int kb = 0; kb < SEQ; kb += 128) {
    // K tile 128x64 via TDM (overlaps with V-transpose staging below)
    if (w == 0) tdm_load_tile64(lds_base + KS_OFF, Kb + (size_t)kb * DKDIM, 128);
    // V tile transposed (1024 chunks of 8, all 4 waves)
#pragma unroll
    for (int i = 0; i < 8; ++i) {
      int c = tid + i * 128;
      int row = c >> 3, cc = (c & 7) * 8;
      bf16x8 vv = *(const bf16x8*)(Vb + (size_t)(kb + row) * DKDIM + cc);
#pragma unroll
      for (int e = 0; e < 8; ++e) Vt[(cc + e) * 132 + row] = vv[e];
    }
    if (w == 0) __builtin_amdgcn_s_wait_tensorcnt(0);
    __syncthreads();

    // Stage A: S(64x128) = Q @ K^T, scaled, into Ss
    {
      v8f sc[2][4] = {};
#pragma unroll
      for (int kk = 0; kk < DKDIM; kk += 32) {
        v16bf a0 = load_frag_a(Qs + (wm * 32 + 0)  * 72 + kk, 72, r, h);
        v16bf a1 = load_frag_a(Qs + (wm * 32 + 16) * 72 + kk, 72, r, h);
#pragma unroll
        for (int fn = 0; fn < 4; ++fn) {
          v16bf bb = load_frag_b(Ks + (wn * 64 + fn * 16) * 72 + kk, 72, r, h);
          sc[0][fn] = wmma_bf16(a0, bb, sc[0][fn]);
          sc[1][fn] = wmma_bf16(a1, bb, sc[1][fn]);
        }
      }
#pragma unroll
      for (int fm = 0; fm < 2; ++fm)
#pragma unroll
        for (int fn = 0; fn < 4; ++fn)
#pragma unroll
          for (int g = 0; g < 8; ++g) {
            int row = wm * 32 + fm * 16 + g + 8 * h;
            int col = wn * 64 + fn * 16 + r;
            Ss[row * 129 + col] = sc[fm][fn][g] * inv_scale;
          }
    }
    __syncthreads();

    // Stage B: online softmax, one row per thread (waves 0-1 fully active)
    if (tid < 64) {
      float* srow = Ss + tid * 129;
      float mloc = -3.0e38f;
#pragma unroll 8
      for (int j = 0; j < 128; ++j) mloc = fmaxf(mloc, srow[j]);
      float mold = m_run[tid];
      float mnew = fmaxf(mold, mloc);
      float alpha = __expf(mold - mnew);
      float lsum = 0.0f;
      bf16* prow = Ps + tid * 132;
#pragma unroll 8
      for (int j = 0; j < 128; ++j) {
        float p = __expf(srow[j] - mnew);
        lsum += p;
        prow[j] = (bf16)p;
      }
      m_run[tid] = mnew;
      l_run[tid] = l_run[tid] * alpha + lsum;
      float* orow = Os + tid * 65;
#pragma unroll 8
      for (int j = 0; j < 64; ++j) orow[j] *= alpha;
    }
    __syncthreads();

    // Stage C: Os(64x64) += P(64x128) @ V(128x64); wave tile 16x64
    {
      v8f oc[4] = {};
#pragma unroll
      for (int kk = 0; kk < 128; kk += 32) {
        v16bf a = load_frag_a(Ps + (w * 16) * 132 + kk, 132, r, h);
#pragma unroll
        for (int fn = 0; fn < 4; ++fn) {
          v16bf bb = load_frag_b(Vt + (fn * 16) * 132 + kk, 132, r, h);
          oc[fn] = wmma_bf16(a, bb, oc[fn]);
        }
      }
#pragma unroll
      for (int fn = 0; fn < 4; ++fn)
#pragma unroll
        for (int g = 0; g < 8; ++g) {
          int row = w * 16 + g + 8 * h;
          int col = fn * 16 + r;
          Os[row * 65 + col] += oc[fn][g];
        }
    }
    __syncthreads();
  }

  // Epilogue: normalize, write merged-head layout [B][S][H*64] as bf16
  const int hh = pair >> 1, b = pair & 1;
#pragma unroll
  for (int i = 0; i < 32; ++i) {
    int idx = tid + i * 128;                   // 4096 elems
    int m = idx >> 6, d = idx & 63;
    float val = Os[m * 65 + d] / l_run[m];
    Ob[((size_t)(b * SEQ + s0 + m)) * DM + hh * DKDIM + d] = (bf16)val;
  }
}

// ---------------------------------------------------------------------------
// Kernel 4: output GEMM  out = O(bf16) @ Wo(bf16,T) + bo  -> f32
// grid = (DM/128, M/128), 256 threads
// ---------------------------------------------------------------------------
__global__ __launch_bounds__(256) void mha_outproj_kernel(
    const bf16* __restrict__ Ob, const bf16* __restrict__ Wot,
    const float* __restrict__ bo, float* __restrict__ out) {
  __shared__ bf16 As[128 * 72];
  __shared__ bf16 Bs[128 * 72];

  const int tid = threadIdx.x;
  const int lane = tid & 31, w = tid >> 5;
  const int r = lane & 15, h = lane >> 4;
  const int wm = w >> 1, wn = w & 1;
  const int n0 = blockIdx.x * 128;
  const int m0 = blockIdx.y * 128;

  v8f acc[2][4] = {};

  for (int k0 = 0; k0 < DM; k0 += 64) {
    if (k0 + 64 < DM) {
      __builtin_prefetch(Ob + (m0 + (tid >> 1)) * DM + k0 + 64, 0, 1);
      __builtin_prefetch(Wot + (n0 + (tid >> 1)) * DM + k0 + 64, 0, 1);
    }
#pragma unroll
    for (int i = 0; i < 4; ++i) {
      int c = tid + i * 256;
      int row = c >> 3, cc = (c & 7) * 8;
      *(bf16x8*)(As + row * 72 + cc) =
          *(const bf16x8*)(Ob + (size_t)(m0 + row) * DM + k0 + cc);
      *(bf16x8*)(Bs + row * 72 + cc) =
          *(const bf16x8*)(Wot + (size_t)(n0 + row) * DM + k0 + cc);
    }
    __syncthreads();

#pragma unroll
    for (int kk = 0; kk < 64; kk += 32) {
      v16bf a0 = load_frag_a(As + (wm * 32 + 0)  * 72 + kk, 72, r, h);
      v16bf a1 = load_frag_a(As + (wm * 32 + 16) * 72 + kk, 72, r, h);
#pragma unroll
      for (int fn = 0; fn < 4; ++fn) {
        v16bf bb = load_frag_b(Bs + (wn * 64 + fn * 16) * 72 + kk, 72, r, h);
        acc[0][fn] = wmma_bf16(a0, bb, acc[0][fn]);
        acc[1][fn] = wmma_bf16(a1, bb, acc[1][fn]);
      }
    }
    __syncthreads();
  }

#pragma unroll
  for (int fm = 0; fm < 2; ++fm) {
#pragma unroll
    for (int fn = 0; fn < 4; ++fn) {
      int n = n0 + wn * 64 + fn * 16 + r;
      float bias = bo[n];
#pragma unroll
      for (int g = 0; g < 8; ++g) {
        int m = m0 + wm * 32 + fm * 16 + g + 8 * h;
        out[(size_t)m * DM + n] = acc[fm][fn][g] + bias;
      }
    }
  }
}

// ---------------------------------------------------------------------------
extern "C" void kernel_launch(void* const* d_in, const int* in_sizes, int n_in,
                              void* d_out, int out_size, void* d_ws, size_t ws_size,
                              hipStream_t stream) {
  const float* q  = (const float*)d_in[0];
  const float* k  = (const float*)d_in[1];
  const float* v  = (const float*)d_in[2];
  const float* Wq = (const float*)d_in[3];
  const float* bq = (const float*)d_in[4];
  const float* Wo = (const float*)d_in[5];
  const float* bo = (const float*)d_in[6];
  float* out = (float*)d_out;

  // Workspace layout (bf16 elements): Wqt(1M) Wot(1M) Qp(4M) Kp(4M) Vp(4M) Ob(4M)
  bf16* Wqt = (bf16*)d_ws;
  bf16* Wot = Wqt + (size_t)1024 * 1024;
  bf16* Qp  = Wot + (size_t)1024 * 1024;
  bf16* Kp  = Qp  + (size_t)HEADS * BATCH * SEQ * DKDIM;
  bf16* Vp  = Kp  + (size_t)HEADS * BATCH * SEQ * DKDIM;
  bf16* Ob  = Vp  + (size_t)HEADS * BATCH * SEQ * DKDIM;

  // 1) weights -> bf16 transposed
  mha_convert_weights<<<(2 * 1024 * 1024) / 256, 256, 0, stream>>>(Wq, Wo, Wqt, Wot);

  // 2) Q/K/V projections (all use Wq/bq, faithful to reference)
  dim3 pgrid(DM / 128, (BATCH * SEQ) / 128, 3);
  mha_proj_kernel<<<pgrid, 256, 0, stream>>>(q, k, v, Wqt, bq, Qp, Kp, Vp);

  // 3) flash attention per (head, batch, q-tile)
  dim3 fgrid(SEQ / 64, HEADS * BATCH);
  mha_flash_kernel<<<fgrid, 128, FLASH_LDS_BYTES, stream>>>(Qp, Kp, Vp, Ob);

  // 4) output projection
  dim3 ogrid(DM / 128, (BATCH * SEQ) / 128);
  mha_outproj_kernel<<<ogrid, 256, 0, stream>>>(Ob, Wot, bo, out);
}
